// VQuantLinear_4690104287252
// MI455X (gfx1250) — matrix operand order repaired and checked by reading, not compiled, not run on hardware
//
#include <hip/hip_runtime.h>
#include <hip/hip_bf16.h>

// ---------------------------------------------------------------------------
// Feature probes (compile-safe): async global->LDS copies on gfx1250
// ---------------------------------------------------------------------------
#if defined(__has_builtin)
#  if __has_builtin(__builtin_amdgcn_global_load_async_to_lds_b128)
#    define USE_ASYNC_LDS 1
#  endif
#endif
#ifndef USE_ASYNC_LDS
#  define USE_ASYNC_LDS 0
#endif

#if USE_ASYNC_LDS
#  if __has_builtin(__builtin_amdgcn_s_wait_asynccnt)
#    define WAIT_ASYNC0() __builtin_amdgcn_s_wait_asynccnt(0)
#  else
#    define WAIT_ASYNC0() asm volatile("s_wait_asynccnt 0" ::: "memory")
#  endif
#endif

typedef __attribute__((ext_vector_type(16))) _Float16 v16h;
typedef __attribute__((ext_vector_type(8)))  _Float16 v8h;
typedef __attribute__((ext_vector_type(8)))  float    v8f;
typedef __attribute__((ext_vector_type(4)))  int      v4i;

typedef __attribute__((address_space(1))) v4i* gptr_v4i;   // global (prints as __device__)
typedef __attribute__((address_space(3))) v4i* lptr_v4i;   // LDS    (prints as __shared__)

#define IN_F    4096
#define OUT_F   11008
#define TOKENS  8192

#define BM 128
#define BN 256
#define BK 32
#define LDA 40                 // padded LDS row stride in halves (80B): 16-lane
                               // strided b128 reads hit all 64 banks disjointly
#define NKT (IN_F / BK)        // 128 k-steps

#if USE_ASYNC_LDS
__device__ __forceinline__ void async_b128(const _Float16* g, _Float16* l) {
    __builtin_amdgcn_global_load_async_to_lds_b128(
        (gptr_v4i)g, (lptr_v4i)l, 0, 0);
}
#endif

// ---------------------------------------------------------------------------
// Pass 0: convert X f32 -> f16 into workspace
// ---------------------------------------------------------------------------
__global__ __launch_bounds__(256)
void cvt_x_f16_kernel(const float* __restrict__ x, _Float16* __restrict__ x16) {
    int i = (blockIdx.x * 256 + threadIdx.x) * 4;   // total = 2^25, exact
    float4 v = *(const float4*)(x + i);
    union { _Float16 h[4]; uint2 u; } t;
    t.h[0] = (_Float16)v.x; t.h[1] = (_Float16)v.y;
    t.h[2] = (_Float16)v.z; t.h[3] = (_Float16)v.w;
    *(uint2*)(x16 + i) = t.u;
}

// ---------------------------------------------------------------------------
// Pass 1: dequantize weight into f16 [OUT_F][IN_F], permutation applied on rows
//   w[o][k] = (cent[idx[o/8][k]][o%8] + rcent[ridx[o/8][k]][o%8]) * scale[o] + bias[o]
//   W16[perm[o]][k] = w[o][k]      (== w[inv_perm] in the reference)
// ---------------------------------------------------------------------------
__global__ __launch_bounds__(256)
void dequant_w_kernel(const float* __restrict__ cent,
                      const float* __restrict__ rcent,
                      const float* __restrict__ scale,
                      const float* __restrict__ bias,
                      const int*   __restrict__ idx,
                      const int*   __restrict__ ridx,
                      const int*   __restrict__ perm,
                      _Float16*    __restrict__ w16) {
    long t = (long)blockIdx.x * 256 + threadIdx.x;     // over OUT_F*IN_F (exact)
    int  k = (int)(t & (IN_F - 1));
    int  o = (int)(t >> 12);                           // IN_F == 2^12
    int  v = o >> 3;
    int  l = o & 7;
    int  ci = idx [v * IN_F + k];
    int  ri = ridx[v * IN_F + k];
    float val = cent[ci * 8 + l] + rcent[ri * 8 + l];
    val = val * scale[o] + bias[o];
    int of = perm[o];
    w16[(long)of * IN_F + k] = (_Float16)val;
}

// ---------------------------------------------------------------------------
// Pass 2: C[TOKENS, OUT_F] = A(f16) x B with Bmem = W16[OUT_F][IN_F] (K-contig rows,
// as read into this kernel's B fragment: lane = N column, 16 contiguous K halves per half-wave).
// 256 threads = 8 waves in 2(M) x 4(N); each wave owns a 64x64 tile = 4x4 WMMA tiles
// (16 v_wmma per k-step per wave, 8 fragment loads).
// Tiles stream HBM/L2 -> LDS via GLOBAL_LOAD_ASYNC_TO_LDS_B128 (ASYNCcnt),
// double-buffered, one s_wait_asynccnt + barrier per k-step.
// ---------------------------------------------------------------------------
__global__ __launch_bounds__(256)
void gemm_wmma_kernel(const _Float16* __restrict__ A,
                      const _Float16* __restrict__ B,
                      float* __restrict__ C) {
    __shared__ _Float16 ldsA[2][BM * LDA];   // 10 KB per buffer
    __shared__ _Float16 ldsB[2][BN * LDA];   // 20 KB per buffer

    const int tid   = threadIdx.x;
    const int lane  = tid & 31;
    const int wave  = tid >> 5;
    const int waveM = wave >> 2;             // 0..1
    const int waveN = wave & 3;              // 0..3
    const int m0    = blockIdx.y * BM;
    const int n0    = blockIdx.x * BN;

    // per-thread 16B copy chunks: A tile = 512 chunks (2/thread), B = 1024 (4/thread)
    const int rA0 = tid >> 2;                // 0..63
    const int rA1 = (tid + 256) >> 2;        // 64..127
    const int kA  = (tid & 3) * 8;
    int rB[4];
#pragma unroll
    for (int q = 0; q < 4; ++q) rB[q] = (tid + 256 * q) >> 2;   // 0..255

    const _Float16* gA = A + (long)m0 * IN_F;
    const _Float16* gB = B + (long)n0 * IN_F;

    // fragment indexing (wave32; A and B both use contiguous k = 16*(lane/16) + s)
    const int mlane = lane & 15;
    const int kselA = (lane < 16) ? 0 : 16;  // A: 16 contiguous halves
    const int kselB = (lane < 16) ? 0 : 16;  // B: 16 contiguous halves

    v8f acc[4][4] = {};

    // ---- preload k-tile 0 ----
#if USE_ASYNC_LDS
    async_b128(gA + (long)rA0 * IN_F + kA, &ldsA[0][rA0 * LDA + kA]);
    async_b128(gA + (long)rA1 * IN_F + kA, &ldsA[0][rA1 * LDA + kA]);
#pragma unroll
    for (int q = 0; q < 4; ++q)
        async_b128(gB + (long)rB[q] * IN_F + kA, &ldsB[0][rB[q] * LDA + kA]);
    WAIT_ASYNC0();
#else
    {
        uint4 a0 = *(const uint4*)(gA + (long)rA0 * IN_F + kA);
        uint4 a1 = *(const uint4*)(gA + (long)rA1 * IN_F + kA);
        uint4 b0 = *(const uint4*)(gB + (long)rB[0] * IN_F + kA);
        uint4 b1 = *(const uint4*)(gB + (long)rB[1] * IN_F + kA);
        uint4 b2 = *(const uint4*)(gB + (long)rB[2] * IN_F + kA);
        uint4 b3 = *(const uint4*)(gB + (long)rB[3] * IN_F + kA);
        *(uint4*)(&ldsA[0][rA0 * LDA + kA])   = a0;
        *(uint4*)(&ldsA[0][rA1 * LDA + kA])   = a1;
        *(uint4*)(&ldsB[0][rB[0] * LDA + kA]) = b0;
        *(uint4*)(&ldsB[0][rB[1] * LDA + kA]) = b1;
        *(uint4*)(&ldsB[0][rB[2] * LDA + kA]) = b2;
        *(uint4*)(&ldsB[0][rB[3] * LDA + kA]) = b3;
    }
#endif
    __syncthreads();

    for (int kt = 0; kt < NKT; ++kt) {
        const int cur  = kt & 1;
        const int nxt  = cur ^ 1;
        const bool more = (kt + 1) < NKT;

#if USE_ASYNC_LDS
        // kick off next tile's DMA into the other buffer, then compute
        if (more) {
            const _Float16* pA = gA + (kt + 1) * BK;
            const _Float16* pB = gB + (kt + 1) * BK;
            async_b128(pA + (long)rA0 * IN_F + kA, &ldsA[nxt][rA0 * LDA + kA]);
            async_b128(pA + (long)rA1 * IN_F + kA, &ldsA[nxt][rA1 * LDA + kA]);
#pragma unroll
            for (int q = 0; q < 4; ++q)
                async_b128(pB + (long)rB[q] * IN_F + kA, &ldsB[nxt][rB[q] * LDA + kA]);
        }
#else
        uint4 na0, na1, nb[4];
        if (more) {
            const _Float16* pA = gA + (kt + 1) * BK;
            const _Float16* pB = gB + (kt + 1) * BK;
            na0 = *(const uint4*)(pA + (long)rA0 * IN_F + kA);
            na1 = *(const uint4*)(pA + (long)rA1 * IN_F + kA);
#pragma unroll
            for (int q = 0; q < 4; ++q)
                nb[q] = *(const uint4*)(pB + (long)rB[q] * IN_F + kA);
        }
#endif

        // ---- compute on current LDS buffer: 16 WMMAs per wave ----
        const _Float16* sA = &ldsA[cur][0];
        const _Float16* sB = &ldsB[cur][0];

        v16h bfrag[4];
#pragma unroll
        for (int j = 0; j < 4; ++j) {
            int nrow = waveN * 64 + j * 16 + mlane;
            bfrag[j] = *(const v16h*)(sB + nrow * LDA + kselB);
        }
#pragma unroll
        for (int i = 0; i < 4; ++i) {
            int arow = waveM * 64 + i * 16 + mlane;
            v8h lo = *(const v8h*)(sA + arow * LDA + kselA);
            v8h hi = *(const v8h*)(sA + arow * LDA + kselA + 8);
            v16h afrag = __builtin_shufflevector(lo, hi,
                0, 1, 2, 3, 4, 5, 6, 7, 8, 9, 10, 11, 12, 13, 14, 15);
#pragma unroll
            for (int j = 0; j < 4; ++j) {
                acc[i][j] = __builtin_amdgcn_wmma_f32_16x16x32_f16(
                    false, afrag, false, bfrag[j],
                    (short)0, acc[i][j], false, false);
            }
        }

#if USE_ASYNC_LDS
        if (more) WAIT_ASYNC0();
#else
        if (more) {
            *(uint4*)(&ldsA[nxt][rA0 * LDA + kA])   = na0;
            *(uint4*)(&ldsA[nxt][rA1 * LDA + kA])   = na1;
#pragma unroll
            for (int q = 0; q < 4; ++q)
                *(uint4*)(&ldsB[nxt][rB[q] * LDA + kA]) = nb[q];
        }
#endif
        __syncthreads();
    }

    // ---- store C (D-matrix layout: lane<16 -> M rows 0..7, lane>=16 -> 8..15) ----
#pragma unroll
    for (int i = 0; i < 4; ++i) {
#pragma unroll
        for (int j = 0; j < 4; ++j) {
            int mrow = m0 + waveM * 64 + i * 16 + ((lane < 16) ? 0 : 8);
            int ncol = n0 + waveN * 64 + j * 16 + mlane;
            float* p = C + (long)mrow * OUT_F + ncol;
#pragma unroll
            for (int r = 0; r < 8; ++r) {
                p[(long)r * OUT_F] = acc[i][j][r];
            }
        }
    }
}

// ---------------------------------------------------------------------------
extern "C" void kernel_launch(void* const* d_in, const int* in_sizes, int n_in,
                              void* d_out, int out_size, void* d_ws, size_t ws_size,
                              hipStream_t stream) {
    const float* x     = (const float*)d_in[0];
    const float* cent  = (const float*)d_in[1];
    const float* rcent = (const float*)d_in[2];
    const float* scale = (const float*)d_in[3];
    const float* bias  = (const float*)d_in[4];
    const int*   idx   = (const int*)d_in[5];
    const int*   ridx  = (const int*)d_in[6];
    const int*   perm  = (const int*)d_in[7];
    float* out = (float*)d_out;

    _Float16* x16 = (_Float16*)d_ws;                                      // 67 MB
    _Float16* w16 = (_Float16*)((char*)d_ws + (size_t)TOKENS * IN_F * 2); // +90 MB

    // Pass 0: X -> f16
    {
        int blocks = (TOKENS * IN_F) / (4 * 256);   // 32768
        cvt_x_f16_kernel<<<blocks, 256, 0, stream>>>(x, x16);
    }
    // Pass 1: dequant + scale/bias + permutation scatter -> W16[OUT][IN]
    {
        long total = (long)OUT_F * IN_F;            // 45,088,768
        int blocks = (int)(total / 256);            // 176128, exact
        dequant_w_kernel<<<blocks, 256, 0, stream>>>(cent, rcent, scale, bias,
                                                     idx, ridx, perm, w16);
    }
    // Pass 2: WMMA GEMM
    {
        dim3 grid(OUT_F / BN, TOKENS / BM);         // (43, 64)
        gemm_wmma_kernel<<<grid, 256, 0, stream>>>(x16, w16, out);
    }
}
